// SANet_62534723830348
// MI455X (gfx1250) — matrix operand order repaired and checked by reading, not compiled, not run on hardware
//
#include <hip/hip_runtime.h>
#include <hip/hip_bf16.h>

typedef __attribute__((ext_vector_type(16))) _Float16 v16h;
typedef __attribute__((ext_vector_type(8)))  float    v8f;
typedef __attribute__((ext_vector_type(2)))  _Float16 h2;

#define HW   4096
#define NCH  64
#define BK   64
#define KPAD 72   // f16 stride padding for K/V tiles
#define OPAD 68   // f32 stride padding for O staging

__global__ __launch_bounds__(256) void sanet_attn_kernel(
    const float* __restrict__ content,   // Fc  [b, c, HW]  (queries, also copied to out)
    const float* __restrict__ fs,        // Fs  [b, c, HW]  (keys)
    const float* __restrict__ fst,       // Fst [b, c, HW]  (values)
    float* __restrict__ out)             // [b, 2c, HW]
{
    __shared__ _Float16 ldsK[BK * KPAD];     // [kk][ch]  (pairs along ch contiguous)
    __shared__ _Float16 ldsV[NCH * KPAD];    // [ch][kk]  (pairs along kk contiguous)
    __shared__ float    ldsO[8][16 * OPAD];  // per-wave staging (also reused for P as f16)

    const int tid   = threadIdx.x;
    const int lane  = tid & 31;
    const int wave  = tid >> 5;
    const int nlo   = lane & 15;     // N index within fragment / low lane id
    const int khalf = lane >> 4;     // lane-half selector in A/B layouts

    const int b   = blockIdx.x >> 5;         // 32 q-blocks per batch
    const int q0  = (blockIdx.x & 31) * 128; // 128 q rows per WG
    const int q0w = q0 + wave * 16;          // 16 q rows per wave

    const float* cbase = content + (size_t)b * NCH * HW;
    const float* kbase = fs      + (size_t)b * NCH * HW;
    const float* vbase = fst     + (size_t)b * NCH * HW;
    float*       obase = out     + (size_t)b * (2 * NCH) * HW;

    // ---- pass-through copy: out[:, 0:64, :] = content (coalesced along q) ----
    #pragma unroll 4
    for (int i = 0; i < 32; ++i) {
        int idx = tid + 256 * i;            // 64ch * 128q = 8192
        int ch  = idx >> 7;
        int q   = q0 + (idx & 127);
        obase[ch * HW + q] = cbase[ch * HW + q];
    }

    // ---- load Q fragments (A layout, 16x32 f16, 2 frags cover ch 0..63) ----
    v16h aq[2];
    {
        const int m = q0w + nlo;
        #pragma unroll
        for (int f = 0; f < 2; ++f) {
            #pragma unroll
            for (int j = 0; j < 8; ++j) {
                int ch = f * 32 + (j >> 2) * 16 + khalf * 8 + (j & 3) * 2;
                aq[f][2 * j]     = (_Float16)cbase[ch * HW + m];
                aq[f][2 * j + 1] = (_Float16)cbase[(ch + 1) * HW + m];
            }
        }
    }

    // ---- online-softmax state (C-layout: lane owns rows m = khalf*8 + j) ----
    v8f oacc[4];
    float mrun[8], lrun[8];
    #pragma unroll
    for (int j = 0; j < 8; ++j) { mrun[j] = -3.0e38f; lrun[j] = 0.f; }
    #pragma unroll
    for (int t = 0; t < 4; ++t)
        #pragma unroll
        for (int j = 0; j < 8; ++j) oacc[t][j] = 0.f;

    _Float16* ldsPw = (_Float16*)(&ldsO[wave][0]);  // 16 x BK f16, stride KPAD

    for (int k0 = 0; k0 < HW; k0 += BK) {
        __syncthreads();  // protect K/V slabs from previous iteration's readers

        // ---- cooperative stage K,V tiles (global reads coalesced along kk) ----
        #pragma unroll 4
        for (int i = 0; i < 16; ++i) {
            int idx = tid + 256 * i;        // 64ch * 64kk = 4096
            int kk  = idx & 63;
            int ch  = idx >> 6;
            float kv = kbase[ch * HW + k0 + kk];
            float vv = vbase[ch * HW + k0 + kk];
            ldsK[kk * KPAD + ch] = (_Float16)kv;   // transposed: [kk][ch]
            ldsV[ch * KPAD + kk] = (_Float16)vv;   // direct:     [ch][kk]
        }
        if (k0 + BK < HW)
            __builtin_prefetch(kbase + k0 + BK + (tid << 4), 0, 1);
        __syncthreads();

        // ---- S tile: 16q x 64k via 8 WMMAs (2 per 16-col subtile) ----
        v8f sf[4];
        #pragma unroll
        for (int t = 0; t < 4; ++t) {
            v16h bk0, bk1;
            const int kkcol = t * 16 + nlo;   // S column (key index)
            #pragma unroll
            for (int j = 0; j < 8; ++j) {
                int ch0 = (j >> 2) * 16 + khalf * 8 + (j & 3) * 2; // contraction rows 0..31
                h2 p0 = *(const h2*)&ldsK[kkcol * KPAD + ch0];
                h2 p1 = *(const h2*)&ldsK[kkcol * KPAD + 32 + ch0];
                bk0[2 * j] = p0.x; bk0[2 * j + 1] = p0.y;
                bk1[2 * j] = p1.x; bk1[2 * j + 1] = p1.y;
            }
            v8f z;
            #pragma unroll
            for (int j = 0; j < 8; ++j) z[j] = 0.f;
            v8f acc = __builtin_amdgcn_wmma_f32_16x16x32_f16(
                false, aq[0], false, bk0, (short)0, z, false, false);
            sf[t] = __builtin_amdgcn_wmma_f32_16x16x32_f16(
                false, aq[1], false, bk1, (short)0, acc, false, false);
        }

        // ---- online softmax over this 64-wide tile ----
        float tmax[8];
        #pragma unroll
        for (int j = 0; j < 8; ++j)
            tmax[j] = fmaxf(fmaxf(sf[0][j], sf[1][j]), fmaxf(sf[2][j], sf[3][j]));
        #pragma unroll
        for (int d = 1; d < 16; d <<= 1)
            #pragma unroll
            for (int j = 0; j < 8; ++j)
                tmax[j] = fmaxf(tmax[j], __shfl_xor(tmax[j], d, 16));

        float psum[8];
        #pragma unroll
        for (int j = 0; j < 8; ++j) {
            float mnew  = fmaxf(mrun[j], tmax[j]);
            float alpha = __expf(mrun[j] - mnew);
            mrun[j] = mnew;
            lrun[j] *= alpha;
            #pragma unroll
            for (int t = 0; t < 4; ++t) oacc[t][j] *= alpha;
            float s = 0.f;
            #pragma unroll
            for (int t = 0; t < 4; ++t) {
                float p = __expf(sf[t][j] - mnew);
                sf[t][j] = p;
                s += p;
            }
            psum[j] = s;
        }
        #pragma unroll
        for (int d = 1; d < 16; d <<= 1)
            #pragma unroll
            for (int j = 0; j < 8; ++j)
                psum[j] += __shfl_xor(psum[j], d, 16);
        #pragma unroll
        for (int j = 0; j < 8; ++j) lrun[j] += psum[j];

        // ---- C-layout P -> A-layout via per-wave LDS (same-wave LDS is in-order) ----
        #pragma unroll
        for (int t = 0; t < 4; ++t)
            #pragma unroll
            for (int j = 0; j < 8; ++j)
                ldsPw[(khalf * 8 + j) * KPAD + t * 16 + nlo] = (_Float16)sf[t][j];

        v16h ap[2];
        #pragma unroll
        for (int f = 0; f < 2; ++f)
            #pragma unroll
            for (int j = 0; j < 8; ++j) {
                int kk = f * 32 + (j >> 2) * 16 + khalf * 8 + (j & 3) * 2;
                h2 p = *(const h2*)&ldsPw[nlo * KPAD + kk];
                ap[f][2 * j] = p.x; ap[f][2 * j + 1] = p.y;
            }

        // ---- O += P * V^T : 8 WMMAs over 4 channel tiles ----
        #pragma unroll
        for (int t = 0; t < 4; ++t) {
            v16h bv0, bv1;
            const int chcol = t * 16 + nlo;   // output channel
            #pragma unroll
            for (int j = 0; j < 8; ++j) {
                int kk0 = (j >> 2) * 16 + khalf * 8 + (j & 3) * 2; // contraction rows (kk) 0..31
                h2 p0 = *(const h2*)&ldsV[chcol * KPAD + kk0];
                h2 p1 = *(const h2*)&ldsV[chcol * KPAD + 32 + kk0];
                bv0[2 * j] = p0.x; bv0[2 * j + 1] = p0.y;
                bv1[2 * j] = p1.x; bv1[2 * j + 1] = p1.y;
            }
            oacc[t] = __builtin_amdgcn_wmma_f32_16x16x32_f16(
                false, ap[0], false, bv0, (short)0, oacc[t], false, false);
            oacc[t] = __builtin_amdgcn_wmma_f32_16x16x32_f16(
                false, ap[1], false, bv1, (short)0, oacc[t], false, false);
        }
    }

    // ---- normalize and store O (transpose via LDS for q-contiguous stores) ----
    __syncthreads();
    float* ldsOw = &ldsO[wave][0];
    {
        float inv[8];
        #pragma unroll
        for (int j = 0; j < 8; ++j) inv[j] = 1.f / lrun[j];
        #pragma unroll
        for (int t = 0; t < 4; ++t)
            #pragma unroll
            for (int j = 0; j < 8; ++j)
                ldsOw[(khalf * 8 + j) * OPAD + t * 16 + nlo] = oacc[t][j] * inv[j];
    }
    // same-wave LDS is in-order; each wave reads only its own slab
    #pragma unroll 4
    for (int r = 0; r < 32; ++r) {
        int n = 2 * r + khalf;   // channel 0..63
        int m = nlo;             // q offset 0..15
        obase[(NCH + n) * HW + q0w + m] = ldsOw[m * OPAD + n];
    }
}

extern "C" void kernel_launch(void* const* d_in, const int* in_sizes, int n_in,
                              void* d_out, int out_size, void* d_ws, size_t ws_size,
                              hipStream_t stream) {
    (void)in_sizes; (void)n_in; (void)d_ws; (void)ws_size; (void)out_size;
    const float* content = (const float*)d_in[0];
    const float* fs      = (const float*)d_in[1];
    const float* fst     = (const float*)d_in[2];
    float*       out     = (float*)d_out;
    // grid: 4 batches * (4096 / 128 q-rows per WG) = 128 workgroups of 256 threads
    sanet_attn_kernel<<<dim3(128), dim3(256), 0, stream>>>(content, fs, fst, out);
}
